// RecurrentHead_12472585027726
// MI455X (gfx1250) — matrix operand order repaired and checked by reading, not compiled, not run on hardware
//
#include <hip/hip_runtime.h>

#define T_  512
#define B_  256
#define IN_ 256
#define H_  256
#define G3_ 768
#define LDA 264   // bf16 elements per staged A row (256 + 8 pad -> conflict-friendly)
#define LDH 260   // f32 elements per hfp row (pad breaks 0-mod-64 stride)

typedef __attribute__((ext_vector_type(16))) __bf16 v16bf;
typedef __attribute__((ext_vector_type(8)))  float  v8f;

union FragBF { float4 f4[2]; v16bf v; };

// ---------------- zero-fill d_out (unmasked scatter targets must be 0) ----------------
__global__ void zero_f4(float4* __restrict__ p, int n4) {
  int i = blockIdx.x * blockDim.x + threadIdx.x;
  if (i < n4) p[i] = make_float4(0.f, 0.f, 0.f, 0.f);
}

// ---------------- weights fp32 -> bf16 (once; stays L2-resident afterwards) ----------------
__global__ void cvt_bf16(const float* __restrict__ wih, const float* __restrict__ whh,
                         __bf16* __restrict__ wihb, __bf16* __restrict__ whhb) {
  int i = blockIdx.x * blockDim.x + threadIdx.x;
  if (i < G3_ * IN_) { wihb[i] = (__bf16)wih[i]; whhb[i] = (__bf16)whh[i]; }
}

// ---------------- pack/scatter index tables (single workgroup, one-time) ----------------
// orderM[k] = flat index of k-th True of batch_mask (row-major).
// Gin[t*B+b]  = x-row feeding compact[t,b]  (orderM[prank_packed]), or -1 if t>=len[b].
// Gout[t*B+b] = output row receiving h_new of step (t,b): orderM[t*B+b] if packed-true
//               and t*B+b < Ntrue, else -1 (reference scatters sflat by literal flat row).
__global__ __launch_bounds__(1024) void setup_idx(const int* __restrict__ mask,
                                                  int* __restrict__ orderM,
                                                  int* __restrict__ Gin,
                                                  int* __restrict__ Gout) {
  __shared__ int Llen[B_];
  __shared__ int Rex[T_ + 1];
  const int tid = threadIdx.x;
  if (tid < B_) {
    int s = 0;
    for (int t = 0; t < T_; ++t) s += (mask[t * B_ + tid] != 0);
    Llen[tid] = s;
  }
  if (tid < T_) {
    int s = 0;
    for (int b = 0; b < B_; ++b) s += (mask[tid * B_ + b] != 0);
    Rex[tid] = s;
  }
  __syncthreads();
  if (tid == 0) {
    int run = 0;
    for (int t = 0; t < T_; ++t) { int c = Rex[t]; Rex[t] = run; run += c; }
    Rex[T_] = run;
  }
  __syncthreads();
  if (tid < T_) {
    int base = Rex[tid];
    for (int b = 0; b < B_; ++b)
      if (mask[tid * B_ + b] != 0) orderM[base++] = tid * B_ + b;
  }
  __syncthreads();
  if (tid < T_) {
    const int t = tid;
    const int N = Rex[T_];
    int S = 0;                                   // packed-Trues in rows < t
    for (int b = 0; b < B_; ++b) S += min(t, Llen[b]);
    int cnt = 0;                                 // packed-Trues earlier in this row
    for (int b = 0; b < B_; ++b) {
      int j = t * B_ + b;
      if (t < Llen[b]) {
        Gin[j] = orderM[S + cnt]; ++cnt;
        Gout[j] = (j < N) ? orderM[j] : -1;
      } else {
        Gin[j] = -1; Gout[j] = -1;
      }
    }
  }
}

// ---------------- persistent fused GRU scan ----------------
// grid = 16 WGs (batch tiles of M=16), block = 256 (8 waves).
// Wave w owns H columns [32w, 32w+32): 12 WMMA accumulators =
//   {r,z,n} x {input-proj, hidden-proj} x 2 column sub-tiles. Gates computed
//   entirely from registers; only h round-trips through LDS each step.
// `tz` is 0 at runtime but opaque to the compiler: weight addresses depend on t,
// so the 96 loop-invariant B-fragment loads cannot be hoisted+spilled to scratch —
// they stay as per-step global_load_b128s that hit the 192MB L2.
__global__ __launch_bounds__(256) void gru_scan(
    const float* __restrict__ x,
    const float* __restrict__ h0,
    const float* __restrict__ bih,
    const float* __restrict__ bhh,
    const __bf16* __restrict__ Wih,
    const __bf16* __restrict__ Whh,
    const int* __restrict__ Gin,
    const int* __restrict__ Gout,
    float* __restrict__ out,
    int tz)
{
  __shared__ __align__(16) __bf16 xbf[16 * LDA];   // x tile, WMMA-A layout rows
  __shared__ __align__(16) __bf16 hbf[16 * LDA];   // h tile (bf16) for WMMA-A
  __shared__ float hfp[16 * LDH];                  // h tile fp32 (master copy)
  __shared__ float sbi[G3_];
  __shared__ float sbh[G3_];
  __shared__ int   gin_s[16];
  __shared__ int   gout_s[16];

  const int tid  = threadIdx.x;
  const int lane = tid & 31;
  const int wave = tid >> 5;          // 0..7
  const int m    = lane & 15;         // A row / B column within 16-tile
  const int kh   = lane >> 4;         // K-half select (0/1)
  const int b0   = blockIdx.x * 16;   // batch tile origin

  for (int i = tid; i < 16 * H_; i += 256) {
    int b = i >> 8, h = i & 255;
    float v = h0[(b0 + b) * H_ + h];
    hfp[b * LDH + h] = v;
    hbf[b * LDA + h] = (__bf16)v;
  }
  for (int i = tid; i < G3_; i += 256) { sbi[i] = bih[i]; sbh[i] = bhh[i]; }

  const v8f vzero = {0.f, 0.f, 0.f, 0.f, 0.f, 0.f, 0.f, 0.f};

  for (int t = 0; t < T_; ++t) {
    // runtime-zero, compile-time-opaque offset: defeats loop-invariant hoisting
    const size_t anti = (size_t)(t & tz);
    const __bf16* WihT = Wih + anti;
    const __bf16* WhhT = Whh + anti;

    __syncthreads();                                   // gates(t-1) done before restage
    // ---- gather phase: stage compact-x tile + step indices ----
    if (tid < 16) {
      gin_s[tid]  = Gin [t * B_ + b0 + tid];
      gout_s[tid] = Gout[t * B_ + b0 + tid];
    }
    for (int b = 0; b < 16; ++b) {
      int g = Gin[t * B_ + b0 + b];                    // uniform -> scalar load
      float v = (g >= 0) ? x[(size_t)g * IN_ + tid] : 0.f;
      xbf[b * LDA + tid] = (__bf16)v;
    }
    __syncthreads();

    // ---- GEMM phase: 12 x v_wmma_f32_16x16x32_bf16 per K-step, K=256 ----
    v8f acc[12];
#pragma unroll
    for (int j = 0; j < 12; ++j) acc[j] = vzero;

#pragma unroll
    for (int k0 = 0; k0 < 256; k0 += 32) {
      FragBF aX, aH;
      {
        const __bf16* ap = &xbf[m * LDA + k0 + kh * 8];
        aX.f4[0] = *(const float4*)(ap);
        aX.f4[1] = *(const float4*)(ap + 16);
        const __bf16* hp = &hbf[m * LDA + k0 + kh * 8];
        aH.f4[0] = *(const float4*)(hp);
        aH.f4[1] = *(const float4*)(hp + 16);
      }
#pragma unroll
      for (int q = 0; q < 2; ++q) {
#pragma unroll
        for (int s = 0; s < 3; ++s) {
          const int n0 = s * H_ + 32 * wave + 16 * q;  // weight row block (gate s)
          FragBF bI, bH;
          const __bf16* bpI = WihT + (size_t)(n0 + m) * IN_ + k0 + kh * 8;
          bI.f4[0] = *(const float4*)(bpI);
          bI.f4[1] = *(const float4*)(bpI + 16);
          const __bf16* bpH = WhhT + (size_t)(n0 + m) * IN_ + k0 + kh * 8;
          bH.f4[0] = *(const float4*)(bpH);
          bH.f4[1] = *(const float4*)(bpH + 16);
          acc[q * 3 + s] = __builtin_amdgcn_wmma_f32_16x16x32_bf16(
              false, aX.v, false, bI.v, (short)0, acc[q * 3 + s], false, false);
          acc[6 + q * 3 + s] = __builtin_amdgcn_wmma_f32_16x16x32_bf16(
              false, aH.v, false, bH.v, (short)0, acc[6 + q * 3 + s], false, false);
        }
      }
    }
    __syncthreads();                                   // all hbf reads done before update

    // ---- gate phase: straight from WMMA accumulators ----
#pragma unroll
    for (int q = 0; q < 2; ++q) {
      const int hcol = 32 * wave + 16 * q + m;
      const float bir = sbi[hcol],          bhr = sbh[hcol];
      const float biz = sbi[H_ + hcol],     bhz = sbh[H_ + hcol];
      const float bin = sbi[2 * H_ + hcol], bhn = sbh[2 * H_ + hcol];
#pragma unroll
      for (int i = 0; i < 8; ++i) {
        const int row = i + 8 * kh;                    // WMMA C-layout row
        float ir  = acc[q * 3 + 0][i] + bir;
        float iz  = acc[q * 3 + 1][i] + biz;
        float inn = acc[q * 3 + 2][i] + bin;
        float hr  = acc[6 + q * 3 + 0][i] + bhr;
        float hz  = acc[6 + q * 3 + 1][i] + bhz;
        float hn  = acc[6 + q * 3 + 2][i] + bhn;
        float r = 1.f / (1.f + __expf(-(ir + hr)));
        float z = 1.f / (1.f + __expf(-(iz + hz)));
        float n = tanhf(inn + r * hn);
        float hold = hfp[row * LDH + hcol];
        float hnew = (1.f - z) * n + z * hold;
        int gi = gin_s[row];
        int go = gout_s[row];
        float hsel = (gi >= 0) ? hnew : hold;          // freeze h past length
        hfp[row * LDH + hcol] = hsel;
        hbf[row * LDA + hcol] = (__bf16)hsel;
        if (go >= 0) out[(size_t)go * H_ + hcol] = hnew;   // fused masked scatter
      }
    }
  }
  __syncthreads();
  for (int i = tid; i < 16 * H_; i += 256) {           // h_last
    int b = i >> 8, h = i & 255;
    out[(size_t)T_ * B_ * H_ + (size_t)(b0 + b) * H_ + h] = hfp[b * LDH + h];
  }
}

extern "C" void kernel_launch(void* const* d_in, const int* in_sizes, int n_in,
                              void* d_out, int out_size, void* d_ws, size_t ws_size,
                              hipStream_t stream) {
  (void)in_sizes; (void)n_in; (void)ws_size;
  const float* x    = (const float*)d_in[0];
  const float* h0   = (const float*)d_in[1];
  const int*   mask = (const int*)d_in[2];
  const float* Wih  = (const float*)d_in[3];
  const float* Whh  = (const float*)d_in[4];
  const float* bih  = (const float*)d_in[5];
  const float* bhh  = (const float*)d_in[6];
  float* out = (float*)d_out;

  char* ws = (char*)d_ws;
  int*    orderM = (int*)(ws);                         // 512 KB
  int*    Gin    = (int*)(ws + 524288);                // 512 KB
  int*    Gout   = (int*)(ws + 1048576);               // 512 KB
  __bf16* wihb   = (__bf16*)(ws + 1572864);            // 384 KB
  __bf16* whhb   = (__bf16*)(ws + 1572864 + 393216);   // 384 KB

  const int n4 = out_size / 4;                         // out_size divisible by 4
  zero_f4<<<(n4 + 255) / 256, 256, 0, stream>>>((float4*)d_out, n4);
  cvt_bf16<<<(G3_ * IN_ + 255) / 256, 256, 0, stream>>>(Wih, Whh, wihb, whhb);
  setup_idx<<<1, 1024, 0, stream>>>(mask, orderM, Gin, Gout);
  gru_scan<<<16, 256, 0, stream>>>(x, h0, bih, bhh, wihb, whhb, Gin, Gout, out, /*tz=*/0);
}
